// LstmLm_50457275793821
// MI455X (gfx1250) — matrix operand order repaired and checked by reading, not compiled, not run on hardware
//
#include <hip/hip_runtime.h>

// ---------------------------------------------------------------------------
// LSTM LM for MI455X (gfx1250): bf16 WMMA (v_wmma_f32_16x16x32_bf16), wave32.
// ---------------------------------------------------------------------------

typedef __attribute__((ext_vector_type(16))) __bf16 v16bf;
typedef __attribute__((ext_vector_type(8)))  float  v8f;

#define HID    256
#define GATE4  1024
#define SEQLEN 50
#define NBATCH 128
#define NVOCAB 10000
#define MROWS  (NBATCH * SEQLEN)   // 6400 flattened (b,s) rows

// ---- WMMA fragment helpers (layouts per CDNA5 ISA 7.12.2, wave32) ---------

// A fragment: 16x32 bf16 tile of row-major A[M,K] (lda elems).
// lane L: row m0+(L&15); holds K = kb..kb+7 and kb+16..kb+23, kb=(L<16?0:8).
static __device__ __forceinline__ v16bf load_a_frag(const __bf16* __restrict__ A,
                                                    int lda, int m0, int k0) {
    const int lane = threadIdx.x & 31;
    const __bf16* p = A + (m0 + (lane & 15)) * lda + k0 + ((lane >> 4) << 3);
    v16bf a;
#pragma unroll
    for (int j = 0; j < 8; ++j) { a[j] = p[j]; a[j + 8] = p[j + 16]; }
    return a;
}

// B fragment: 32x16 bf16 tile of B[K,N] supplied as row-major B^T[N,K] (ldb elems).
// lane L: column n0+(L&15); holds K = kb..kb+7 and kb+16..kb+23, kb=(L<16?0:8).
static __device__ __forceinline__ v16bf load_b_frag(const __bf16* __restrict__ BT,
                                                    int ldb, int n0, int k0) {
    const int lane = threadIdx.x & 31;
    const __bf16* p = BT + (n0 + (lane & 15)) * ldb + k0 + ((lane >> 4) << 3);
    v16bf b;
#pragma unroll
    for (int j = 0; j < 8; ++j) { b[j] = p[j]; b[j + 8] = p[j + 16]; }
    return b;
}

static __device__ __forceinline__ v8f wmma_bf16(v16bf a, v16bf b, v8f c) {
    return __builtin_amdgcn_wmma_f32_16x16x32_bf16(false, a, false, b,
                                                   (short)0, c, false, false);
}

static __device__ __forceinline__ float sigmoidf_(float x) {
    return 1.0f / (1.0f + __expf(-x));
}

// ---- small conversion / setup kernels -------------------------------------

__global__ void f32_to_bf16_kernel(const float* __restrict__ in,
                                   __bf16* __restrict__ out, int n) {
    for (int i = blockIdx.x * blockDim.x + threadIdx.x; i < n;
         i += gridDim.x * blockDim.x)
        out[i] = (__bf16)in[i];
}

__global__ void bias_sum_kernel(const float* __restrict__ a,
                                const float* __restrict__ b,
                                float* __restrict__ o, int n) {
    int i = blockIdx.x * blockDim.x + threadIdx.x;
    if (i < n) o[i] = a[i] + b[i];
}

// one block per (b,s) row: gather embedding row, cast to bf16
__global__ __launch_bounds__(HID) void embed_gather_kernel(
    const int* __restrict__ x, const float* __restrict__ emb,
    __bf16* __restrict__ xin) {
    const int m   = blockIdx.x;          // m = b*SEQLEN + s
    const int tok = x[m];
    xin[m * HID + threadIdx.x] = (__bf16)emb[tok * HID + threadIdx.x];
}

// ---- input-side GEMM: Y[M,1024] = X[M,256] @ W^T + bias -------------------
// grid (1024/128, M/128), block 256 (8 waves). Wave = one 16-row M tile,
// sweeps 8 N tiles reusing the preloaded A fragments.
__global__ __launch_bounds__(256) void input_gemm_kernel(
    const __bf16* __restrict__ X, const __bf16* __restrict__ WT,
    const float* __restrict__ bias, float* __restrict__ Y) {
    const int wave = threadIdx.x >> 5;
    const int lane = threadIdx.x & 31;
    const int m0   = (blockIdx.y * 8 + wave) * 16;
    const int n0b  = blockIdx.x * 128;

    v16bf afr[8];
#pragma unroll
    for (int kt = 0; kt < 8; ++kt) afr[kt] = load_a_frag(X, HID, m0, kt * 32);

    for (int nt = 0; nt < 8; ++nt) {
        const int n0 = n0b + nt * 16;
        v8f acc = {0.f, 0.f, 0.f, 0.f, 0.f, 0.f, 0.f, 0.f};
#pragma unroll
        for (int kt = 0; kt < 8; ++kt)
            acc = wmma_bf16(afr[kt], load_b_frag(WT, HID, n0, kt * 32), acc);

        const int n  = n0 + (lane & 15);
        const int mb = m0 + ((lane >> 4) << 3);
        const float bv = bias[n];
#pragma unroll
        for (int r = 0; r < 8; ++r) Y[(mb + r) * GATE4 + n] = acc[r] + bv;
    }
}

// ---- recurrent scan: 8 persistent blocks, 16 batch rows each --------------
// h kept in LDS (bf16, WMMA A source); c kept in registers (fp32, per wave:
// 32 hidden units x 16 rows = two v8f fragments). w_hh streams from L2.
__global__ __launch_bounds__(256) void lstm_scan_kernel(
    const float* __restrict__ xg,      // [6400,1024], row m = b*SEQLEN+s
    const __bf16* __restrict__ whhT,   // [1024,256] = (h->gates W)^T layout
    __bf16* __restrict__ hout) {       // [6400,256]
    __shared__ __bf16 h_lds[16 * HID];

    const int wave = threadIdx.x >> 5;
    const int lane = threadIdx.x & 31;
    const int b0   = blockIdx.x * 16;       // batch rows b0..b0+15
    const int j0   = wave * 32;             // hidden units owned by this wave
    const int mb   = (lane >> 4) << 3;      // fragment row base for this lane
    const int nl   = lane & 15;             // fragment col for this lane

    for (int i = threadIdx.x; i < 16 * HID; i += 256) h_lds[i] = (__bf16)0.0f;
    __syncthreads();

    float c_st[2][8];
#pragma unroll
    for (int t = 0; t < 2; ++t)
#pragma unroll
        for (int r = 0; r < 8; ++r) c_st[t][r] = 0.0f;

    for (int s = 0; s < SEQLEN; ++s) {
        // stage A fragments (shared across all 8 gate tiles of this wave)
        v16bf afr[8];
#pragma unroll
        for (int kt = 0; kt < 8; ++kt)
            afr[kt] = load_a_frag(h_lds, HID, 0, kt * 32);
        __syncthreads();   // all reads of h_lds done before this step's writes

        // gates: i,f,g,o  x  two 16-wide j sub-tiles
        v8f gf[4][2];
#pragma unroll
        for (int g = 0; g < 4; ++g) {
#pragma unroll
            for (int t = 0; t < 2; ++t) {
                const int n0 = g * 256 + j0 + t * 16;
                v8f acc = {0.f, 0.f, 0.f, 0.f, 0.f, 0.f, 0.f, 0.f};
#pragma unroll
                for (int kt = 0; kt < 8; ++kt)
                    acc = wmma_bf16(afr[kt],
                                    load_b_frag(whhT, HID, n0, kt * 32), acc);
                const int n = n0 + nl;
#pragma unroll
                for (int r = 0; r < 8; ++r)
                    acc[r] += xg[((b0 + mb + r) * SEQLEN + s) * GATE4 + n];
                gf[g][t] = acc;
            }
        }

        // elementwise gate math + state update + publish h
#pragma unroll
        for (int t = 0; t < 2; ++t) {
            const int j = j0 + t * 16 + nl;
#pragma unroll
            for (int r = 0; r < 8; ++r) {
                const float iv = sigmoidf_(gf[0][t][r]);
                const float fv = sigmoidf_(gf[1][t][r]);
                const float gv = tanhf(gf[2][t][r]);
                const float ov = sigmoidf_(gf[3][t][r]);
                const float cv = fv * c_st[t][r] + iv * gv;
                c_st[t][r] = cv;
                const __bf16 hb = (__bf16)(ov * tanhf(cv));
                const int m = mb + r;
                h_lds[m * HID + j] = hb;
                hout[((b0 + m) * SEQLEN + s) * HID + j] = hb;
            }
        }
        __syncthreads();   // h_lds writes visible before next step's reads
    }
}

// ---- tied-weight projection: Y[6400,10000] = H2 @ emb^T -------------------
// grid (ceil(625/8), 50), block 256. Wave = one 16-row M tile, 8 N tiles.
__global__ __launch_bounds__(256) void proj_gemm_kernel(
    const __bf16* __restrict__ H2, const __bf16* __restrict__ EMB,
    float* __restrict__ Y) {
    const int wave = threadIdx.x >> 5;
    const int lane = threadIdx.x & 31;
    const int m0   = (blockIdx.y * 8 + wave) * 16;
    const int ntb  = blockIdx.x * 8;

    v16bf afr[8];
#pragma unroll
    for (int kt = 0; kt < 8; ++kt) afr[kt] = load_a_frag(H2, HID, m0, kt * 32);

    for (int nt = 0; nt < 8; ++nt) {
        const int ntile = ntb + nt;
        if (ntile >= NVOCAB / 16) break;    // uniform branch: EXEC stays full
        const int n0 = ntile * 16;
        __builtin_prefetch(EMB + (n0 + 16) * HID, 0, 1);   // global_prefetch
        v8f acc = {0.f, 0.f, 0.f, 0.f, 0.f, 0.f, 0.f, 0.f};
#pragma unroll
        for (int kt = 0; kt < 8; ++kt)
            acc = wmma_bf16(afr[kt], load_b_frag(EMB, HID, n0, kt * 32), acc);

        const int n  = n0 + (lane & 15);
        const int mb = m0 + ((lane >> 4) << 3);
#pragma unroll
        for (int r = 0; r < 8; ++r) Y[(mb + r) * NVOCAB + n] = acc[r];
    }
}

// ---------------------------------------------------------------------------

extern "C" void kernel_launch(void* const* d_in, const int* in_sizes, int n_in,
                              void* d_out, int out_size, void* d_ws, size_t ws_size,
                              hipStream_t stream) {
    const int*   x     = (const int*)  d_in[0];
    const float* emb   = (const float*)d_in[1];
    const float* w_ih0 = (const float*)d_in[2];
    const float* w_hh0 = (const float*)d_in[3];
    const float* b_ih0 = (const float*)d_in[4];
    const float* b_hh0 = (const float*)d_in[5];
    const float* w_ih1 = (const float*)d_in[6];
    const float* w_hh1 = (const float*)d_in[7];
    const float* b_ih1 = (const float*)d_in[8];
    const float* b_hh1 = (const float*)d_in[9];
    float* out = (float*)d_out;

    char*  ws  = (char*)d_ws;
    size_t off = 0;
    auto alloc = [&](size_t bytes) -> char* {
        char* p = ws + off;
        off += (bytes + 255) & ~size_t(255);
        return p;
    };
    __bf16* emb_bf = (__bf16*)alloc((size_t)NVOCAB * HID * sizeof(__bf16));
    __bf16* w0i    = (__bf16*)alloc((size_t)GATE4 * HID * sizeof(__bf16));
    __bf16* w0h    = (__bf16*)alloc((size_t)GATE4 * HID * sizeof(__bf16));
    __bf16* w1i    = (__bf16*)alloc((size_t)GATE4 * HID * sizeof(__bf16));
    __bf16* w1h    = (__bf16*)alloc((size_t)GATE4 * HID * sizeof(__bf16));
    float*  bias0  = (float*) alloc((size_t)GATE4 * sizeof(float));
    float*  bias1  = (float*) alloc((size_t)GATE4 * sizeof(float));
    __bf16* xin    = (__bf16*)alloc((size_t)MROWS * HID * sizeof(__bf16));
    __bf16* h1     = (__bf16*)alloc((size_t)MROWS * HID * sizeof(__bf16));
    __bf16* h2     = (__bf16*)alloc((size_t)MROWS * HID * sizeof(__bf16));
    float*  xg     = (float*) alloc((size_t)MROWS * GATE4 * sizeof(float));
    (void)ws_size; (void)in_sizes; (void)n_in; (void)out_size;

    // weight / embedding precision conversion + bias fold
    f32_to_bf16_kernel<<<2048, 256, 0, stream>>>(emb,   emb_bf, NVOCAB * HID);
    f32_to_bf16_kernel<<<512,  256, 0, stream>>>(w_ih0, w0i,    GATE4 * HID);
    f32_to_bf16_kernel<<<512,  256, 0, stream>>>(w_hh0, w0h,    GATE4 * HID);
    f32_to_bf16_kernel<<<512,  256, 0, stream>>>(w_ih1, w1i,    GATE4 * HID);
    f32_to_bf16_kernel<<<512,  256, 0, stream>>>(w_hh1, w1h,    GATE4 * HID);
    bias_sum_kernel<<<4, 256, 0, stream>>>(b_ih0, b_hh0, bias0, GATE4);
    bias_sum_kernel<<<4, 256, 0, stream>>>(b_ih1, b_hh1, bias1, GATE4);

    // embedding gather
    embed_gather_kernel<<<MROWS, HID, 0, stream>>>(x, emb, xin);

    // layer 0: batched input GEMM, then recurrent scan
    input_gemm_kernel<<<dim3(GATE4 / 128, MROWS / 128), 256, 0, stream>>>(
        xin, w0i, bias0, xg);
    lstm_scan_kernel<<<NBATCH / 16, 256, 0, stream>>>(xg, w0h, h1);

    // layer 1 (xg buffer reused)
    input_gemm_kernel<<<dim3(GATE4 / 128, MROWS / 128), 256, 0, stream>>>(
        h1, w1i, bias1, xg);
    lstm_scan_kernel<<<NBATCH / 16, 256, 0, stream>>>(xg, w1h, h2);

    // tied-weight projection (store-bandwidth bound: 256 MB fp32 out)
    proj_gemm_kernel<<<dim3((NVOCAB / 16 + 7) / 8, MROWS / 128), 256, 0,
                       stream>>>(h2, emb_bf, out);
}
